// TokenFeatures_2448131358768
// MI455X (gfx1250) — compile-verified, matrix operand order checked by reading, not run on hardware
//
#include <hip/hip_runtime.h>
#include <hip/hip_bf16.h>
#include <math.h>

// ---------------- problem constants ----------------
#define BB      2
#define NN      1024
#define KK      32
#define NUM_RBF 16
#define NUM_POS 16
#define EDGE_C  128
#define MAXREL  32
#define PP      65
#define EDGE_IN 1057            // NUM_POS + NUM_RBF*PP + 1
#define KPAD    1088            // EDGE_IN padded to 34*32
#define KTILES  34              // KPAD / 32
#define EDGES   (BB*NN*KK)      // 65536
#define MT      32              // edges (M rows) per block in the GEMM kernel

typedef __attribute__((ext_vector_type(16))) _Float16 v16h;
typedef __attribute__((ext_vector_type(8)))  _Float16 v8h;
typedef __attribute__((ext_vector_type(8)))  float    v8f;

// SRC = concat(repeat(arange(5),5), repeat(arange(4),10))
__device__ __constant__ int cSRC[PP] = {
    0,0,0,0,0, 1,1,1,1,1, 2,2,2,2,2, 3,3,3,3,3, 4,4,4,4,4,
    0,0,0,0,0,0,0,0,0,0, 1,1,1,1,1,1,1,1,1,1,
    2,2,2,2,2,2,2,2,2,2, 3,3,3,3,3,3,3,3,3,3 };
// DST = concat(tile(arange(5),5), tile(arange(5,15),4))
__device__ __constant__ int cDST[PP] = {
    0,1,2,3,4, 0,1,2,3,4, 0,1,2,3,4, 0,1,2,3,4, 0,1,2,3,4,
    5,6,7,8,9,10,11,12,13,14, 5,6,7,8,9,10,11,12,13,14,
    5,6,7,8,9,10,11,12,13,14, 5,6,7,8,9,10,11,12,13,14 };

// ---------------------------------------------------------------------------
// Kernel 1: X = coords + noise, mask-condition, build 15-atom set (with Cb)
// atoms15 layout: [B*N][15][3] f32
// ---------------------------------------------------------------------------
__global__ __launch_bounds__(256) void atoms_kernel(
    const float* __restrict__ coords, const float* __restrict__ mask,
    const float* __restrict__ noise, float* __restrict__ atoms15)
{
    int ri = blockIdx.x * blockDim.x + threadIdx.x;
    if (ri >= BB * NN) return;
    const float* c  = coords + (size_t)ri * 42;
    const float* nz = noise  + (size_t)ri * 42;
    const float* mk = mask   + (size_t)ri * 14;

    float X[14][3];
    float x1[3];
    #pragma unroll
    for (int d = 0; d < 3; ++d) x1[d] = c[3 + d] + nz[3 + d];
    #pragma unroll
    for (int a = 0; a < 14; ++a) {
        bool keep = mk[a] > 0.0f;
        #pragma unroll
        for (int d = 0; d < 3; ++d)
            X[a][d] = keep ? (c[a*3 + d] + nz[a*3 + d]) : x1[d];
    }
    // Cb = -0.58273431*cross(b,c) + 0.56802827*b - 0.54067466*c + X1
    float bv[3], cv[3], av[3], cb[3];
    #pragma unroll
    for (int d = 0; d < 3; ++d) { bv[d] = X[1][d] - X[0][d]; cv[d] = X[2][d] - X[1][d]; }
    av[0] = bv[1]*cv[2] - bv[2]*cv[1];
    av[1] = bv[2]*cv[0] - bv[0]*cv[2];
    av[2] = bv[0]*cv[1] - bv[1]*cv[0];
    #pragma unroll
    for (int d = 0; d < 3; ++d)
        cb[d] = -0.58273431f*av[d] + 0.56802827f*bv[d] - 0.54067466f*cv[d] + X[1][d];

    float* o = atoms15 + (size_t)ri * 45;
    #pragma unroll
    for (int a = 0; a < 4; ++a)
        for (int d = 0; d < 3; ++d) o[a*3 + d] = X[a][d];
    #pragma unroll
    for (int d = 0; d < 3; ++d) o[12 + d] = cb[d];
    #pragma unroll
    for (int a = 4; a < 14; ++a)
        for (int d = 0; d < 3; ++d) o[(a+1)*3 + d] = X[a][d];
}

// ---------------------------------------------------------------------------
// Kernel 2: pack edge_W (f32 [1057][128]) -> f16 WMMA B-operand tiles.
// Layout: half index ((kt*8 + nt)*32 + lane)*16 + v,
//   n = nt*16 + (lane&15)
//   k = kt*32 + koff,  koff = lane<16 ? (v<8 ? v : v+8) : (v<8 ? v+8 : v+16)
// k >= 1057 rows are zero (padding).
// ---------------------------------------------------------------------------
__global__ __launch_bounds__(256) void wpack_kernel(
    const float* __restrict__ edge_W, _Float16* __restrict__ Wpack)
{
    int idx = blockIdx.x * 256 + threadIdx.x;
    if (idx >= KTILES * 8 * 32 * 16) return;
    int v  = idx & 15;
    int l  = (idx >> 4) & 31;
    int nt = (idx >> 9) & 7;
    int kt = idx >> 12;
    int koff = (l < 16) ? ((v < 8) ? v : v + 8) : ((v < 8) ? v + 8 : v + 16);
    int k = kt * 32 + koff;
    int n = nt * 16 + (l & 15);
    float val = (k < EDGE_IN) ? edge_W[(size_t)k * EDGE_C + n] : 0.0f;
    Wpack[idx] = (_Float16)val;
}

// ---------------------------------------------------------------------------
// Kernel 3: per-row CA distances + top-K (ascending, tie -> smaller index)
// One block of 256 threads per (b,i) row.
// ---------------------------------------------------------------------------
__global__ __launch_bounds__(256) void topk_kernel(
    const float* __restrict__ atoms15, float* __restrict__ outEidx,
    float* __restrict__ outD, int* __restrict__ EidxWs)
{
    __shared__ float dist[NN];
    __shared__ float sval[256];
    __shared__ int   sidx[256];

    int bi  = blockIdx.x;            // 0 .. B*N-1
    int b   = bi / NN;
    int tid = threadIdx.x;

    const float* ca = atoms15 + (size_t)bi * 45 + 3;   // atom 1 = CA
    float cx = ca[0], cy = ca[1], cz = ca[2];

    for (int j = tid; j < NN; j += 256) {
        const float* p = atoms15 + ((size_t)(b * NN + j)) * 45 + 3;
        float dx = cx - p[0], dy = cy - p[1], dz = cz - p[2];
        dist[j] = sqrtf(dx*dx + dy*dy + dz*dz + 1e-6f);
    }
    __syncthreads();

    for (int k = 0; k < KK; ++k) {
        float bv = 3.0e38f;
        int   bj = NN;
        for (int j = tid; j < NN; j += 256) {
            float v = dist[j];
            if (v < bv || (v == bv && j < bj)) { bv = v; bj = j; }
        }
        sval[tid] = bv; sidx[tid] = bj;
        __syncthreads();
        for (int s = 128; s > 0; s >>= 1) {
            if (tid < s) {
                float ov = sval[tid + s]; int oj = sidx[tid + s];
                if (ov < sval[tid] || (ov == sval[tid] && oj < sidx[tid])) {
                    sval[tid] = ov; sidx[tid] = oj;
                }
            }
            __syncthreads();
        }
        if (tid == 0) {
            int jm = sidx[0];
            EidxWs[(size_t)bi * KK + k]  = jm;
            outEidx[(size_t)bi * KK + k] = (float)jm;
            outD[(size_t)bi * KK + k]    = sval[0];
            dist[jm] = 3.0e38f;
        }
        __syncthreads();
    }
}

// ---------------------------------------------------------------------------
// Kernel 4: fused edge-feature build + [32 x 1088] x [1088 x 128] WMMA GEMM
//           + cross-wave LayerNorm. One block (256 thr = 8 waves) per 32
//           edges. Each wave owns one 16-col slice of the 128 outputs and
//           processes TWO 16-row A tiles against a single shared B load per
//           k-step (2 WMMAs per B fetch -> halves B traffic per FLOP).
// ---------------------------------------------------------------------------
__global__ __launch_bounds__(256) void edge_gemm_kernel(
    const float* __restrict__ atoms15, const int* __restrict__ Eidx,
    const _Float16* __restrict__ Wpack,
    const int* __restrict__ resIdx, const int* __restrict__ asymId,
    const float* __restrict__ tokenBonds, const int* __restrict__ isLigand,
    const float* __restrict__ posW, const float* __restrict__ posB,
    const float* __restrict__ lnG, const float* __restrict__ lnB,
    float* __restrict__ Eout)
{
    __shared__ __align__(16) _Float16 feat[MT * KPAD];   // 68 KB
    __shared__ float ctile[MT * EDGE_C];                 // 16 KB
    __shared__ float psum[256], psq[256];

    const int tid = threadIdx.x;
    const int blk = blockIdx.x;              // EDGES/MT blocks

    // ---- zero K-padding rows [1057,1088) ----
    for (int t = tid; t < MT * (KPAD - EDGE_IN); t += 256) {
        int e  = t / (KPAD - EDGE_IN);
        int kk = EDGE_IN + t % (KPAD - EDGE_IN);
        feat[e * KPAD + kk] = (_Float16)0.0f;
    }

    // ---- positional (16) + token-bond (1) features: one thread per edge ----
    if (tid < MT) {
        int ge  = blk * MT + tid;
        int b   = ge / (NN * KK);
        int rem = ge % (NN * KK);
        int i   = rem / KK;
        int j   = Eidx[ge];
        int off = resIdx[b * NN + i] - resIdx[b * NN + j];
        int same = (asymId[b * NN + i] == asymId[b * NN + j]);
        int dcl = off + MAXREL;
        dcl = dcl < 0 ? 0 : (dcl > 2 * MAXREL ? 2 * MAXREL : dcl);
        int d = same ? dcl : (2 * MAXREL + 1);
        #pragma unroll
        for (int r = 0; r < NUM_POS; ++r)
            feat[tid * KPAD + r] = (_Float16)(posW[d * NUM_POS + r] + posB[r]);
        int lig = isLigand[b * NN + i] | isLigand[b * NN + j];
        float tb = lig ? tokenBonds[((size_t)(b * NN + i)) * NN + j] : 0.0f;
        feat[tid * KPAD + (EDGE_IN - 1)] = (_Float16)tb;
    }

    // ---- RBF features: MT edges x 65 atom pairs ----
    const float inv_sigma = 1.0f / 1.25f;             // sigma = 20/16
    for (int t = tid; t < MT * PP; t += 256) {
        int e   = t / PP;
        int p   = t % PP;
        int ge  = blk * MT + e;
        int b   = ge / (NN * KK);
        int rem = ge % (NN * KK);
        int i   = rem / KK;
        int j   = Eidx[ge];
        const float* A  = atoms15 + ((size_t)(b * NN + i) * 15 + cSRC[p]) * 3;
        const float* Bm = atoms15 + ((size_t)(b * NN + j) * 15 + cDST[p]) * 3;
        float dx = A[0] - Bm[0], dy = A[1] - Bm[1], dz = A[2] - Bm[2];
        float dval = sqrtf(dx*dx + dy*dy + dz*dz + 1e-6f);
        _Float16* f = &feat[e * KPAD + NUM_POS + p * NUM_RBF];
        #pragma unroll
        for (int r = 0; r < NUM_RBF; ++r) {
            float mu = 2.0f + (float)r * (20.0f / 15.0f);   // linspace(2,22,16)
            float u  = (dval - mu) * inv_sigma;
            f[r] = (_Float16)__expf(-u * u);
        }
    }
    __syncthreads();

    // ---- WMMA GEMM ----
    const int lane = tid & 31;
    const int wave = tid >> 5;                         // nt = 0..7
    const int m    = lane & 15;
    const int chunk = (lane < 16) ? 0 : 8;             // ISA 16-bit A layout
    const _Float16* arow0 = &feat[m * KPAD + chunk];          // rows  0..15
    const _Float16* arow1 = &feat[(m + 16) * KPAD + chunk];   // rows 16..31
    const v16h* bbase = ((const v16h*)Wpack) + (size_t)wave * 32 + lane;

    v8f acc0 = {};
    v8f acc1 = {};
    for (int kt = 0; kt < KTILES; ++kt) {
        v16h bm = bbase[(size_t)kt * 256];              // shared by both tiles

        v8h alo0 = *(const v8h*)(arow0 + kt * 32);      // K {0..7} / {8..15}
        v8h ahi0 = *(const v8h*)(arow0 + kt * 32 + 16); // K {16..23} / {24..31}
        v16h a0 = __builtin_shufflevector(alo0, ahi0,
                  0,1,2,3,4,5,6,7,8,9,10,11,12,13,14,15);
        acc0 = __builtin_amdgcn_wmma_f32_16x16x32_f16(
                   false, a0, false, bm, (short)0, acc0, false, false);

        v8h alo1 = *(const v8h*)(arow1 + kt * 32);
        v8h ahi1 = *(const v8h*)(arow1 + kt * 32 + 16);
        v16h a1 = __builtin_shufflevector(alo1, ahi1,
                  0,1,2,3,4,5,6,7,8,9,10,11,12,13,14,15);
        acc1 = __builtin_amdgcn_wmma_f32_16x16x32_f16(
                   false, a1, false, bm, (short)0, acc1, false, false);
    }

    // ---- spill C to LDS (ISA C layout: VGPR v -> rows v / v+8) ----
    {
        int crow0 = (lane < 16) ? 0 : 8;
        int ccol  = wave * 16 + (lane & 15);
        #pragma unroll
        for (int v = 0; v < 8; ++v) {
            ctile[(crow0 + v) * EDGE_C + ccol]        = acc0[v];
            ctile[(crow0 + v + 16) * EDGE_C + ccol]   = acc1[v];
        }
    }
    __syncthreads();

    // ---- LayerNorm over the 128 outputs of each of the 32 edge rows ----
    int row = tid >> 3;        // 0..31
    int s   = tid & 7;         // 8 threads per row, 16 cols each
    float sm = 0.0f, sq = 0.0f;
    #pragma unroll
    for (int c = 0; c < 16; ++c) {
        float v = ctile[row * EDGE_C + s * 16 + c];
        sm += v; sq += v * v;
    }
    psum[tid] = sm; psq[tid] = sq;
    __syncthreads();
    for (int st = 4; st > 0; st >>= 1) {
        if (s < st) { psum[tid] += psum[tid + st]; psq[tid] += psq[tid + st]; }
        __syncthreads();
    }
    float mu  = psum[row * 8] * (1.0f / EDGE_C);
    float var = psq[row * 8] * (1.0f / EDGE_C) - mu * mu;
    float inv = rsqrtf(var + 1e-5f);

    float* out = Eout + ((size_t)blk * MT + row) * EDGE_C;
    #pragma unroll
    for (int c = 0; c < 16; ++c) {
        int col = s * 16 + c;
        float v = ctile[row * EDGE_C + col];
        out[col] = (v - mu) * inv * lnG[col] + lnB[col];
    }
}

// ---------------------------------------------------------------------------
extern "C" void kernel_launch(void* const* d_in, const int* in_sizes, int n_in,
                              void* d_out, int out_size, void* d_ws, size_t ws_size,
                              hipStream_t stream)
{
    (void)in_sizes; (void)n_in; (void)out_size; (void)ws_size;
    const float* coords = (const float*)d_in[0];
    const float* mask   = (const float*)d_in[1];
    const float* noise  = (const float*)d_in[2];
    const int*   resIdx = (const int*)d_in[3];
    const int*   asymId = (const int*)d_in[4];
    const float* tBonds = (const float*)d_in[5];
    const int*   isLig  = (const int*)d_in[6];
    const float* posW   = (const float*)d_in[7];
    const float* posB   = (const float*)d_in[8];
    const float* edgeW  = (const float*)d_in[9];
    const float* lnG    = (const float*)d_in[10];
    const float* lnB    = (const float*)d_in[11];

    float* out = (float*)d_out;
    float* outE    = out;                                     // B*N*K*128
    float* outEidx = out + (size_t)EDGES * EDGE_C;            // B*N*K
    float* outD    = outEidx + EDGES;                         // B*N*K

    char* ws = (char*)d_ws;
    float*     atoms15 = (float*)ws;                          // 368,640 B
    int*       EidxWs  = (int*)(ws + 368640);                 // 262,144 B
    _Float16*  Wpack   = (_Float16*)(ws + 368640 + 262144);   // 278,528 B

    atoms_kernel<<<(BB * NN + 255) / 256, 256, 0, stream>>>(coords, mask, noise, atoms15);
    wpack_kernel<<<(KTILES * 8 * 32 * 16 + 255) / 256, 256, 0, stream>>>(edgeW, Wpack);
    topk_kernel<<<BB * NN, 256, 0, stream>>>(atoms15, outEidx, outD, EidxWs);
    edge_gemm_kernel<<<EDGES / MT, 256, 0, stream>>>(
        atoms15, EidxWs, Wpack, resIdx, asymId, tBonds, isLig,
        posW, posB, lnG, lnB, outE);
}
